// RPN_18090402251363
// MI455X (gfx1250) — matrix operand order repaired and checked by reading, not compiled, not run on hardware
//
#include <hip/hip_runtime.h>
#include <hip/hip_bf16.h>
#include <math.h>

typedef __bf16 bfloat;
typedef __bf16 v16bf __attribute__((ext_vector_type(16)));
typedef float  v8f   __attribute__((ext_vector_type(8)));

#define HWF       1850      // 37*50 feature map
#define NBOX      22200     // 12 * 1850
#define NPAD      32768     // bitonic pad
#define PRE_TOPN  6000
#define POST_TOPN 300
#define NMSW      94        // ceil(6000/64)
#define NSUB      8         // spatial sub-tiles per wave (16 px each)
#define ZBN       544       // zero dummy-tile elements (>= 512+16+16)

// ---------------------------------------------------------------------------
// Zero-fill small buffer (dummy B tile for padded/out-of-range loads)
// ---------------------------------------------------------------------------
__global__ void zfill(bfloat* __restrict__ z, int n) {
  int i = blockIdx.x * blockDim.x + threadIdx.x;
  if (i < n) z[i] = (bfloat)0.0f;
}

// ---------------------------------------------------------------------------
// Weight transform: f32 OIHW (Cout,Cin,3,3) -> bf16 [dyx][Cout][Cin]
// ---------------------------------------------------------------------------
__global__ void wxform(const float* __restrict__ w, bfloat* __restrict__ wt,
                       int Cout, int Cin, int n) {
  int idx = blockIdx.x * blockDim.x + threadIdx.x;
  if (idx >= n) return;
  int dyx = idx % 9;
  int t   = idx / 9;
  int ci  = t % Cin;
  int o   = t / Cin;
  wt[((size_t)dyx * Cout + o) * Cin + ci] = (bfloat)w[idx];
}

// ---------------------------------------------------------------------------
// Conv1: 3->64, f32 NCHW input -> bf16 NHWC output, ReLU (0.5% of FLOPs)
// ---------------------------------------------------------------------------
__global__ void conv1_scalar(const float* __restrict__ img, const float* __restrict__ w,
                             const float* __restrict__ b, bfloat* __restrict__ out,
                             int H, int W) {
  int idx = blockIdx.x * blockDim.x + threadIdx.x;
  int total = H * W * 64;
  if (idx >= total) return;
  int o = idx & 63;
  int p = idx >> 6;
  int y = p / W, x = p % W;
  float s = b[o];
  for (int ci = 0; ci < 3; ci++)
    for (int dy = -1; dy <= 1; dy++)
      for (int dx = -1; dx <= 1; dx++) {
        int iy = y + dy, ix = x + dx;
        if ((unsigned)iy < (unsigned)H && (unsigned)ix < (unsigned)W)
          s += img[((size_t)ci * H + iy) * W + ix] *
               w[((o * 3 + ci) * 3 + (dy + 1)) * 3 + (dx + 1)];
      }
  out[(size_t)p * 64 + o] = (bfloat)fmaxf(s, 0.0f);
}

// ---------------------------------------------------------------------------
// Implicit-GEMM 3x3 conv, pad 1, stride 1.  bf16 NHWC in/out + ReLU.
// One wave: 16(Cout) x 128(spatial) = 8 accumulators.
// Block: 4 waves -> 64(Cout) x 128(spatial).
// A = weights [dyx][Cout][Cin] (16x32 bf16 tile), B = im2col (32x16 bf16).
// Out-of-range B tiles read a zeroed dummy buffer -> branch-free K loop.
// D = A*B + C via v_wmma_f32_16x16x32_bf16, f32 accumulation.
// ---------------------------------------------------------------------------
__global__ __launch_bounds__(128)
void conv3x3_wmma(const bfloat* __restrict__ in, bfloat* __restrict__ out,
                  const bfloat* __restrict__ wt, const float* __restrict__ bias,
                  const bfloat* __restrict__ zerobuf,
                  int Cin, int Cout, int H, int W, float* __restrict__ f32out) {
  const int HW   = H * W;
  const int lane = threadIdx.x & 31;
  const int wv   = threadIdx.x >> 5;
  const int nl   = lane & 15;
  const int kh   = lane >> 4;                   // K-half selector (0/1)
  const int m0   = blockIdx.y * 64 + wv * 16;   // Cout tile base
  const int n0   = blockIdx.x * (16 * NSUB);    // spatial tile base

  int py[NSUB], px[NSUB];
  bool pv[NSUB];
#pragma unroll
  for (int j = 0; j < NSUB; j++) {
    int p = n0 + j * 16 + nl;
    pv[j] = p < HW;
    int pc = pv[j] ? p : 0;
    py[j] = pc / W;
    px[j] = pc % W;
  }

  v8f acc[NSUB];
#pragma unroll
  for (int j = 0; j < NSUB; j++)
#pragma unroll
    for (int r = 0; r < 8; r++) acc[j][r] = 0.0f;

  const int arow = m0 + nl;  // A-matrix row = output channel
  const bfloat* zb = zerobuf + kh * 8;
  for (int dyx = 0; dyx < 9; dyx++) {
    const int dy = dyx / 3 - 1, dx = dyx % 3 - 1;
    const bfloat* wp = wt + ((size_t)dyx * Cout + arow) * Cin + kh * 8;
    const bfloat* bp[NSUB];
#pragma unroll
    for (int j = 0; j < NSUB; j++) {
      int iy = py[j] + dy, ix = px[j] + dx;
      bool bv = pv[j] && (unsigned)iy < (unsigned)H && (unsigned)ix < (unsigned)W;
      bp[j] = bv ? (in + ((size_t)iy * W + ix) * Cin + kh * 8) : zb;
    }
    for (int c0 = 0; c0 < Cin; c0 += 32) {
      __builtin_prefetch(wp + c0 + 32, 0, 3);  // -> global_prefetch_b8
      union { v16bf v; uint4 q[2]; } a;
      // A layout (16x32 bf16): lane<16: elems0-7 = K 0..7, elems8-15 = K 16..23
      //                        lane>=16: K ranges +8  (kh folded into wp)
      a.q[0] = *(const uint4*)(wp + c0);
      a.q[1] = *(const uint4*)(wp + c0 + 16);
#pragma unroll
      for (int j = 0; j < NSUB; j++) {
        union { v16bf v; uint4 q[2]; } bm;
        // B layout (32x16 bf16): col = lane&15, same K order as A (elems0-7 = K 0..7, elems8-15 = K 16..23, +8 for lane>=16)
        bm.q[0] = *(const uint4*)(bp[j] + c0);
        bm.q[1] = *(const uint4*)(bp[j] + c0 + 16);
        acc[j] = __builtin_amdgcn_wmma_f32_16x16x32_bf16(
            false, a.v, false, bm.v, (short)0, acc[j], false, false);
      }
    }
  }

  // D layout: row = r + 8*kh, col = nl
#pragma unroll
  for (int j = 0; j < NSUB; j++) {
    int p = n0 + j * 16 + nl;
    if (p < HW) {
#pragma unroll
      for (int r = 0; r < 8; r++) {
        int m = m0 + kh * 8 + r;
        float v = fmaxf(acc[j][r] + bias[m], 0.0f);
        out[(size_t)p * Cout + m] = (bfloat)v;
        if (f32out) f32out[(size_t)m * HW + p] = v;
      }
    }
  }
}

// ---------------------------------------------------------------------------
// 2x2 maxpool stride 2 (VALID), bf16 NHWC
// ---------------------------------------------------------------------------
__global__ void maxpool2(const bfloat* __restrict__ in, bfloat* __restrict__ out,
                         int H, int W, int C) {
  int Ho = H / 2, Wo = W / 2;
  int idx = blockIdx.x * blockDim.x + threadIdx.x;
  int total = Ho * Wo * C;
  if (idx >= total) return;
  int c = idx % C;
  int p = idx / C;
  int yo = p / Wo, xo = p % Wo;
  const bfloat* r0 = in + ((size_t)(2 * yo) * W + 2 * xo) * C + c;
  const bfloat* r1 = r0 + (size_t)W * C;
  float m = fmaxf(fmaxf((float)r0[0], (float)r0[C]),
                  fmaxf((float)r1[0], (float)r1[C]));
  out[(size_t)p * C + c] = (bfloat)m;
}

// ---------------------------------------------------------------------------
// 1x1 conv heads: bf16 NHWC in, f32 (Cout,512,1,1) weights -> f32 NCHW out
// ---------------------------------------------------------------------------
__global__ void conv1x1(const bfloat* __restrict__ in, const float* __restrict__ w,
                        const float* __restrict__ b, float* __restrict__ out,
                        int Cout, int HW, int Cin) {
  int idx = blockIdx.x * blockDim.x + threadIdx.x;
  if (idx >= Cout * HW) return;
  int o = idx / HW;
  int p = idx % HW;
  const bfloat* ip = in + (size_t)p * Cin;
  const float* wp = w + (size_t)o * Cin;
  float s = b[o];
  for (int c = 0; c < Cin; c++) s += (float)ip[c] * wp[c];
  out[(size_t)o * HW + p] = s;
}

// ---------------------------------------------------------------------------
// Softmax over channel pairs (c, c+12), matching view(B,2,-1,W) softmax(dim=1)
// ---------------------------------------------------------------------------
__global__ void softmax12(const float* __restrict__ cls, float* __restrict__ prob) {
  int idx = blockIdx.x * blockDim.x + threadIdx.x;
  if (idx >= 12 * HWF) return;
  int a = idx / HWF, p = idx % HWF;
  float s0 = cls[(size_t)a * HWF + p];
  float s1 = cls[(size_t)(a + 12) * HWF + p];
  float m = fmaxf(s0, s1);
  float e0 = expf(s0 - m), e1 = expf(s1 - m);
  float d = e0 + e1;
  prob[(size_t)a * HWF + p] = e0 / d;
  prob[(size_t)(a + 12) * HWF + p] = e1 / d;
}

// ---------------------------------------------------------------------------
// Proposal boxes: anchors (generated in-thread), bbox_transform_inv, clip,
// min-size mask (score := -1 when invalid)
// ---------------------------------------------------------------------------
__global__ void proposal_boxes(const float* __restrict__ prob,
                               const float* __restrict__ bbox,
                               const float* __restrict__ iminfo,
                               float* __restrict__ boxes,
                               float* __restrict__ scores) {
  int i = blockIdx.x * blockDim.x + threadIdx.x;
  if (i >= NBOX) return;
  int a = i % 12, p = i / 12;
  int gh = p / 50, gw = p % 50;
  int ri = a >> 2, si = a & 3;
  float ratio = (ri == 0) ? 0.5f : ((ri == 1) ? 1.0f : 2.0f);
  float scale = (float)(4 << si);
  float ws0 = rintf(sqrtf(256.0f / ratio));   // np.round == RNE
  float hs0 = rintf(ws0 * ratio);
  float ws = ws0 * scale, hs = hs0 * scale;
  float sx = (float)(gw * 16), sy = (float)(gh * 16);
  float ax1 = sx + 7.5f - 0.5f * (ws - 1.0f);
  float ay1 = sy + 7.5f - 0.5f * (hs - 1.0f);
  float ax2 = sx + 7.5f + 0.5f * (ws - 1.0f);
  float ay2 = sy + 7.5f + 0.5f * (hs - 1.0f);
  float aw = ax2 - ax1 + 1.0f, ah = ay2 - ay1 + 1.0f;
  float acx = ax1 + 0.5f * aw, acy = ay1 + 0.5f * ah;
  float d0 = bbox[(size_t)(4 * a + 0) * HWF + p];
  float d1 = bbox[(size_t)(4 * a + 1) * HWF + p];
  float d2 = bbox[(size_t)(4 * a + 2) * HWF + p];
  float d3 = bbox[(size_t)(4 * a + 3) * HWF + p];
  float pcx = d0 * aw + acx, pcy = d1 * ah + acy;
  float pw = expf(d2) * aw, ph = expf(d3) * ah;
  float im_h = iminfo[0], im_w = iminfo[1], sc = iminfo[2];
  float x1 = fminf(fmaxf(pcx - 0.5f * pw, 0.0f), im_w - 1.0f);
  float y1 = fminf(fmaxf(pcy - 0.5f * ph, 0.0f), im_h - 1.0f);
  float x2 = fminf(fmaxf(pcx + 0.5f * pw, 0.0f), im_w - 1.0f);
  float y2 = fminf(fmaxf(pcy + 0.5f * ph, 0.0f), im_h - 1.0f);
  float s = prob[(size_t)(12 + a) * HWF + p];
  float minsz = 16.0f * sc;
  bool valid = ((x2 - x1 + 1.0f) >= minsz) && ((y2 - y1 + 1.0f) >= minsz);
  boxes[i * 4 + 0] = x1; boxes[i * 4 + 1] = y1;
  boxes[i * 4 + 2] = x2; boxes[i * 4 + 3] = y2;
  scores[i] = valid ? s : -1.0f;
}

// ---------------------------------------------------------------------------
// Top-k via full bitonic sort of packed u64 keys (score desc, index asc)
// ---------------------------------------------------------------------------
__global__ void key_init(const float* __restrict__ scores,
                         unsigned long long* __restrict__ keys) {
  int i = blockIdx.x * blockDim.x + threadIdx.x;
  if (i >= NPAD) return;
  if (i < NBOX) {
    unsigned u = __float_as_uint(scores[i]);
    u = (u & 0x80000000u) ? ~u : (u | 0x80000000u);  // order-preserving map
    keys[i] = ((unsigned long long)u << 32) |
              (unsigned long long)(0xFFFFFFFFu - (unsigned)i);
  } else {
    keys[i] = 0ull;  // below any real entry
  }
}

__global__ void bitonic_step(unsigned long long* __restrict__ k, int j, int kk) {
  int i = blockIdx.x * blockDim.x + threadIdx.x;
  int p = i ^ j;
  if (p > i) {
    unsigned long long a = k[i], b = k[p];
    bool desc = ((i & kk) == 0);
    if ((a < b) == desc) { k[i] = b; k[p] = a; }
  }
}

__global__ void gather_topk(const unsigned long long* __restrict__ keys,
                            const float* __restrict__ boxes,
                            const float* __restrict__ scores,
                            float* __restrict__ tb, float* __restrict__ tsc,
                            float* __restrict__ area) {
  int i = blockIdx.x * blockDim.x + threadIdx.x;
  if (i >= PRE_TOPN) return;
  unsigned idx = 0xFFFFFFFFu - (unsigned)(keys[i] & 0xFFFFFFFFull);
  float x1 = boxes[idx * 4 + 0], y1 = boxes[idx * 4 + 1];
  float x2 = boxes[idx * 4 + 2], y2 = boxes[idx * 4 + 3];
  tb[i * 4 + 0] = x1; tb[i * 4 + 1] = y1;
  tb[i * 4 + 2] = x2; tb[i * 4 + 3] = y2;
  tsc[i] = scores[idx];
  area[i] = (x2 - x1 + 1.0f) * (y2 - y1 + 1.0f);
}

// ---------------------------------------------------------------------------
// NMS: bitmask build + single-block serial scan (LDS removed-mask)
// ---------------------------------------------------------------------------
__global__ void nms_mask(const float* __restrict__ tb, const float* __restrict__ area,
                         unsigned long long* __restrict__ mask) {
  int i = blockIdx.x * 64 + threadIdx.x;
  int cb = blockIdx.y;
  if (i >= PRE_TOPN) return;
  float x1 = tb[i * 4], y1 = tb[i * 4 + 1], x2 = tb[i * 4 + 2], y2 = tb[i * 4 + 3];
  float ai = area[i];
  unsigned long long bits = 0ull;
  int j0 = cb * 64;
  for (int jj = 0; jj < 64; jj++) {
    int j = j0 + jj;
    if (j < PRE_TOPN && j > i) {
      float xx1 = fmaxf(x1, tb[j * 4]);
      float yy1 = fmaxf(y1, tb[j * 4 + 1]);
      float xx2 = fminf(x2, tb[j * 4 + 2]);
      float yy2 = fminf(y2, tb[j * 4 + 3]);
      float iw = fmaxf(xx2 - xx1 + 1.0f, 0.0f);
      float ih = fmaxf(yy2 - yy1 + 1.0f, 0.0f);
      float inter = iw * ih;
      float iou = inter / (ai + area[j] - inter);
      if (iou > 0.7f) bits |= (1ull << jj);
    }
  }
  mask[(size_t)i * NMSW + cb] = bits;
}

__global__ void nms_scan(const unsigned long long* __restrict__ mask,
                         const float* __restrict__ tsc, int* __restrict__ keep) {
  __shared__ unsigned long long remv[NMSW];
  int t = threadIdx.x;
  if (t < NMSW) remv[t] = 0ull;
  __syncthreads();
  for (int i = 0; i < PRE_TOPN; i++) {
    bool removed = (remv[i >> 6] >> (i & 63)) & 1ull;
    __syncthreads();
    if (!removed && t < NMSW) remv[t] |= mask[(size_t)i * NMSW + t];
    if (t == 0) keep[i] = (!removed && tsc[i] > -1.0f) ? 1 : 0;
    __syncthreads();
  }
}

__global__ void rois_fill(const int* __restrict__ keep, const float* __restrict__ tb,
                          float* __restrict__ rois) {
  if (threadIdx.x != 0 || blockIdx.x != 0) return;
  int cnt = 0;
  for (int i = 0; i < PRE_TOPN && cnt < POST_TOPN; i++) {
    if (keep[i]) {
      rois[cnt * 5 + 0] = 0.0f;
      rois[cnt * 5 + 1] = tb[i * 4 + 0];
      rois[cnt * 5 + 2] = tb[i * 4 + 1];
      rois[cnt * 5 + 3] = tb[i * 4 + 2];
      rois[cnt * 5 + 4] = tb[i * 4 + 3];
      cnt++;
    }
  }
  for (; cnt < POST_TOPN; cnt++)
    for (int j = 0; j < 5; j++) rois[cnt * 5 + j] = 0.0f;
}

// ---------------------------------------------------------------------------
// Host orchestration
// ---------------------------------------------------------------------------
extern "C" void kernel_launch(void* const* d_in, const int* in_sizes, int n_in,
                              void* d_out, int out_size, void* d_ws, size_t ws_size,
                              hipStream_t stream) {
  (void)in_sizes; (void)n_in; (void)out_size; (void)ws_size;
  const float* im_data = (const float*)d_in[0];
  const float* im_info = (const float*)d_in[1];
  const float* vw[13];
  const float* vb[13];
  for (int l = 0; l < 13; l++) {
    vw[l] = (const float*)d_in[2 + 2 * l];
    vb[l] = (const float*)d_in[3 + 2 * l];
  }
  const float* rpn_w   = (const float*)d_in[28];
  const float* rpn_b   = (const float*)d_in[29];
  const float* score_w = (const float*)d_in[30];
  const float* score_b = (const float*)d_in[31];
  const float* bbox_w  = (const float*)d_in[32];
  const float* bbox_b  = (const float*)d_in[33];

  float* feats_out = (float*)d_out;                 // (512,37,50)
  float* rois_out  = feats_out + (size_t)512 * HWF; // (300,5)
  float* prob_out  = rois_out + POST_TOPN * 5;      // (24,37,50)
  float* bbox_out  = prob_out + (size_t)24 * HWF;   // (48,37,50)

  char* wsp = (char*)d_ws;
  auto alloc = [&](size_t bytes) -> char* {
    char* p = wsp;
    wsp += (bytes + 255) & ~(size_t)255;
    return p;
  };

  static const int CIN[13]  = {3, 64, 64, 128, 128, 256, 256, 256, 512, 512, 512, 512, 512};
  static const int COUT[13] = {64, 64, 128, 128, 256, 256, 256, 512, 512, 512, 512, 512, 512};

  bfloat* wt[13];
  wt[0] = nullptr;
  for (int l = 1; l < 13; l++)
    wt[l] = (bfloat*)alloc((size_t)CIN[l] * COUT[l] * 9 * sizeof(bfloat));
  bfloat* wt_rpn  = (bfloat*)alloc((size_t)512 * 512 * 9 * sizeof(bfloat));
  bfloat* zerobuf = (bfloat*)alloc((size_t)ZBN * sizeof(bfloat));
  bfloat* actA = (bfloat*)alloc((size_t)30720000 * sizeof(bfloat)); // 64*600*800
  bfloat* actB = (bfloat*)alloc((size_t)30720000 * sizeof(bfloat));
  float* cls    = (float*)alloc((size_t)24 * HWF * sizeof(float));
  float* boxes  = (float*)alloc((size_t)NBOX * 4 * sizeof(float));
  float* scores = (float*)alloc((size_t)NBOX * sizeof(float));
  unsigned long long* keys = (unsigned long long*)alloc((size_t)NPAD * 8);
  float* tb   = (float*)alloc((size_t)PRE_TOPN * 4 * sizeof(float));
  float* tsc  = (float*)alloc((size_t)PRE_TOPN * sizeof(float));
  float* area = (float*)alloc((size_t)PRE_TOPN * sizeof(float));
  unsigned long long* mask = (unsigned long long*)alloc((size_t)PRE_TOPN * NMSW * 8);
  int* keep = (int*)alloc((size_t)PRE_TOPN * sizeof(int));

  // Zero dummy tile + weight transforms (f32 OIHW -> bf16 [dyx][O][I])
  zfill<<<(ZBN + 255) / 256, 256, 0, stream>>>(zerobuf, ZBN);
  for (int l = 1; l < 13; l++) {
    int n = COUT[l] * CIN[l] * 9;
    wxform<<<(n + 255) / 256, 256, 0, stream>>>(vw[l], wt[l], COUT[l], CIN[l], n);
  }
  {
    int n = 512 * 512 * 9;
    wxform<<<(n + 255) / 256, 256, 0, stream>>>(rpn_w, wt_rpn, 512, 512, n);
  }

  // Conv1 (3->64) scalar
  {
    int H = 600, W = 800, n = H * W * 64;
    conv1_scalar<<<(n + 255) / 256, 256, 0, stream>>>(im_data, vw[0], vb[0], actA, H, W);
  }

  // VGG convs 2..13 with pools after conv indices 1,3,6,9 (0-based)
  bfloat* cur = actA;
  bfloat* nxt = actB;
  int H = 600, W = 800;
  const int SPT = 16 * NSUB;  // spatial per block
  for (int l = 1; l < 13; l++) {
    float* f32o = (l == 12) ? feats_out : nullptr;
    dim3 g((H * W + SPT - 1) / SPT, COUT[l] / 64);
    conv3x3_wmma<<<g, 128, 0, stream>>>(cur, nxt, wt[l], vb[l], zerobuf,
                                        CIN[l], COUT[l], H, W, f32o);
    { bfloat* t = cur; cur = nxt; nxt = t; }
    if (l == 1 || l == 3 || l == 6 || l == 9) {
      int Ho = H / 2, Wo = W / 2, C = COUT[l];
      int n = Ho * Wo * C;
      maxpool2<<<(n + 255) / 256, 256, 0, stream>>>(cur, nxt, H, W, C);
      { bfloat* t = cur; cur = nxt; nxt = t; }
      H = Ho; W = Wo;
    }
  }

  // RPN 3x3 conv (512->512) + ReLU
  {
    dim3 g((H * W + SPT - 1) / SPT, 512 / 64);
    conv3x3_wmma<<<g, 128, 0, stream>>>(cur, nxt, wt_rpn, rpn_b, zerobuf,
                                        512, 512, H, W, nullptr);
    bfloat* t = cur; cur = nxt; nxt = t;
  }

  // 1x1 heads
  conv1x1<<<(24 * HWF + 255) / 256, 256, 0, stream>>>(cur, score_w, score_b, cls, 24, HWF, 512);
  conv1x1<<<(48 * HWF + 255) / 256, 256, 0, stream>>>(cur, bbox_w, bbox_b, bbox_out, 48, HWF, 512);

  // Softmax + proposals
  softmax12<<<(12 * HWF + 255) / 256, 256, 0, stream>>>(cls, prob_out);
  proposal_boxes<<<(NBOX + 255) / 256, 256, 0, stream>>>(prob_out, bbox_out, im_info,
                                                         boxes, scores);

  // Top-6000 via bitonic sort (descending, stable on index)
  key_init<<<NPAD / 256, 256, 0, stream>>>(scores, keys);
  for (int kk = 2; kk <= NPAD; kk <<= 1)
    for (int j = kk >> 1; j > 0; j >>= 1)
      bitonic_step<<<NPAD / 256, 256, 0, stream>>>(keys, j, kk);
  gather_topk<<<(PRE_TOPN + 255) / 256, 256, 0, stream>>>(keys, boxes, scores, tb, tsc, area);

  // NMS + rois
  nms_mask<<<dim3(NMSW, NMSW), 64, 0, stream>>>(tb, area, mask);
  nms_scan<<<1, 128, 0, stream>>>(mask, tsc, keep);
  rois_fill<<<1, 32, 0, stream>>>(keep, tb, rois_out);
}
